// Decoder_53334903881908
// MI455X (gfx1250) — matrix operand order repaired and checked by reading, not compiled, not run on hardware
//
#include <hip/hip_runtime.h>

typedef __attribute__((ext_vector_type(16))) _Float16 v16h;
typedef __attribute__((ext_vector_type(8)))  float    v8f;
typedef __attribute__((ext_vector_type(8)))  unsigned v8u;
typedef __attribute__((ext_vector_type(2)))  _Float16 v2h;

#define DEV __device__ __forceinline__

DEV unsigned pk2h(float lo, float hi) {
  v2h p; p.x = (_Float16)lo; p.y = (_Float16)hi;
  return __builtin_bit_cast(unsigned, p);
}

DEV unsigned bperm(int bpidx, unsigned x) {   // lane^16 swap, index precomputed (bytes)
  return (unsigned)__builtin_amdgcn_ds_bpermute(bpidx, (int)x);
}

DEV float sigm(float x)      { return 1.0f / (1.0f + __expf(-x)); }
DEV float tanh_fast(float x) { return 2.0f / (1.0f + __expf(-2.0f * x)) - 1.0f; }

// Build one 16(M) x 32(K) f16 A-tile in CDNA5 WMMA A layout.
// contiguous k order used here: lane m (0-15) holds row M=m: half j -> K = j + (lane<16 ? 0 : 16), + 32*kc.
// Optional bias column folded at K==bias_k.
DEV v16h make_A(const float* W, int rows, int cols, int mtile, int kc,
                const float* bias1, const float* bias2, int bias_k, int lane) {
  int m   = mtile * 16 + (lane & 15);
  int off = (lane < 16) ? 0 : 16;
  v16h a;
#pragma unroll
  for (int j = 0; j < 16; ++j) {
    int k = kc * 32 + j + off;
    float v = 0.0f;
    if (m < rows) {
      if (k < cols)          v = W[m * cols + k];
      else if (k == bias_k)  v = bias1[m] + (bias2 ? bias2[m] : 0.0f);
    }
    a[j] = (_Float16)v;
  }
  return a;
}

// C/D-layout bias tile: lane n<16 -> (M = mtile*16 + r), lane n>=16 -> (M = mtile*16 + 8 + r).
DEV v8f load_Cbias(const float* bias, int rows, int mtile, int lane) {
  v8f c;
  int mb = mtile * 16 + ((lane < 16) ? 0 : 8);
#pragma unroll
  for (int r = 0; r < 8; ++r) { int m = mb + r; c[r] = (m < rows) ? bias[m] : 0.0f; }
  return c;
}

// Convert two stacked f32 D-tiles (rows 0-15 in d0, rows 16-31 in d1) into the
// f16 B operand, in the contiguous k order make_A uses, for a 32(K) x 16(N) matrix:
//   lanes 0-15 hold K=0-15 of column n, lanes 16-31 hold K=16-31.
// Needs a lane^16 half-swap because D keeps rows 8-15 in the upper lane half.
DEV v16h DD_to_B(v8f d0, v8f d1, int lane, int bpidx) {
  v8u r;
#pragma unroll
  for (int j = 0; j < 4; ++j) {
    unsigned p0 = pk2h(d0[2*j], d0[2*j+1]);
    unsigned p1 = pk2h(d1[2*j], d1[2*j+1]);
    unsigned s0 = bperm(bpidx, p0);
    unsigned s1 = bperm(bpidx, p1);
    r[j]     = (lane < 16) ? p0 : s1;
    r[j + 4] = (lane < 16) ? s0 : p1;
  }
  return __builtin_bit_cast(v16h, r);
}

// D-tile 2 (pose rows 32-47) -> B chunk K=32..63. Row 39 is forced to 1.0 (ones
// column that multiplies the folded bias). K>=40 columns have zero weights, so
// any garbage there (incl. upper-lane contents) is harmless.
DEV v16h D2_to_B(v8f d2, int bpidx) {
  v8u r;
#pragma unroll
  for (int j = 0; j < 4; ++j) {
    float hi   = (j == 3) ? 1.0f : d2[2*j+1];
    unsigned p = pk2h(d2[2*j], hi);
    r[j]     = p;
    r[j + 4] = bperm(bpidx, p);
  }
  return __builtin_bit_cast(v16h, r);
}

DEV v8f wmma(v16h a, v16h b, v8f c) {
  return __builtin_amdgcn_wmma_f32_16x16x32_f16(false, a, false, b, (short)0, c, false, false);
}

DEV void store4(float* p, float a, float b, float c, float d) {
  float t[4] = {a, b, c, d};
  __builtin_memcpy(p, t, 16);  // 4-byte-aligned b128 store
}

// LDS weight tiles: ih 0..15 (mt*2+kc), hh 16..23, mlp 24..26, fc 27..28
#define NT_IH  16
#define NT_HH  8
#define NT_MLP 3
#define NT_FC  2
#define NTILES (NT_IH + NT_HH + NT_MLP + NT_FC)

__global__ __launch_bounds__(256, 4)   // cap at 256 VGPRs, >=4 waves/SIMD
void lstm_decoder_kernel(const float* __restrict__ obs_s,   // [obs_len, B, 39]
                         const float* __restrict__ latent,  // [B, 16]
                         const float* __restrict__ W_ih,    // [128, 39]
                         const float* __restrict__ W_hh,    // [128, 32]
                         const float* __restrict__ b_ih,    // [128]
                         const float* __restrict__ b_hh,    // [128]
                         const float* __restrict__ W_fc,    // [32, 16]
                         const float* __restrict__ b_fc,    // [32]
                         const float* __restrict__ W_mlp,   // [39, 32]
                         const float* __restrict__ b_mlp,   // [39]
                         const int*   __restrict__ pred_len_p,
                         float* __restrict__ out,           // [pred_len, B, 39]
                         int B, int obs_len)
{
  const int lane = threadIdx.x & 31;
  const int wave = threadIdx.x >> 5;
  const int batch_base = (blockIdx.x * 8 + wave) * 16;   // 16 batch rows per wave
  const int bpidx = ((lane ^ 16) << 2);                  // ds_bpermute byte index for lane^16

  // ---- block-cooperative: weights -> WMMA A-layout f16 tiles in LDS ----
  __shared__ v16h wt[NTILES][32];   // 29 KB
  for (int t = wave; t < NTILES; t += 8) {
    v16h a;
    if (t < NT_IH) {
      int mt = t >> 1, kc = t & 1;
      a = make_A(W_ih, 128, 39, mt, kc, b_ih, b_hh, 39, lane);     // K=39 col carries b_ih+b_hh
    } else if (t < NT_IH + NT_HH) {
      a = make_A(W_hh, 128, 32, t - NT_IH, 0, nullptr, nullptr, -1, lane);
    } else if (t < NT_IH + NT_HH + NT_MLP) {
      a = make_A(W_mlp, 39, 32, t - NT_IH - NT_HH, 0, nullptr, nullptr, -1, lane);
    } else {
      a = make_A(W_fc, 32, 16, t - NT_IH - NT_HH - NT_MLP, 0, b_fc, nullptr, 16, lane);  // K=16 col carries b_fc
    }
    wt[t][lane] = a;
  }
  __syncthreads();

  if (batch_base >= B) return;
  const int pred_len = pred_len_p[0];
  const int col = batch_base + (lane & 15);              // this lane's batch column

  v8f c_mlp[3];
#pragma unroll
  for (int mt = 0; mt < 3; ++mt) c_mlp[mt] = load_Cbias(b_mlp, 39, mt, lane);

  // ---- h0^T = W_fc @ latent^T + b_fc (ones column at K=16) ----
  v16h hB;
  {
    const float* lat = latent + (size_t)col * 16;
    v8u r;
#pragma unroll
    for (int j = 0; j < 8; ++j) {
      unsigned hipart = (j == 0) ? pk2h(1.0f, 0.0f) : 0u;    // lanes>=16: K=16 -> 1.0, rest 0
      r[j] = (lane < 16) ? pk2h(lat[2*j], lat[2*j+1]) : hipart;
    }
    v16h latB = __builtin_bit_cast(v16h, r);
    v16h a_fc0 = wt[NT_IH + NT_HH + NT_MLP][lane];
    v16h a_fc1 = wt[NT_IH + NT_HH + NT_MLP + 1][lane];
    v8f z = {};
    v8f h0 = wmma(a_fc0, latB, z);
    v8f h1 = wmma(a_fc1, latB, z);
    hB = DD_to_B(h0, h1, lane, bpidx);
  }

  // ---- last_s = obs_s[-1] -> B-operand chunks (K=0..31, K=32..63 w/ ones at 39) ----
  v16h sB0, sB1;
  {
    const float* src = obs_s + ((size_t)(obs_len - 1) * B + col) * 39;
    int k0 = (lane < 16) ? 0 : 16;
    v8u r;
#pragma unroll
    for (int j = 0; j < 8; ++j) r[j] = pk2h(src[k0 + 2*j], src[k0 + 2*j + 1]);
    sB0 = __builtin_bit_cast(v16h, r);
    v8u r2 = {};
    if (lane < 16) {
      r2[0] = pk2h(src[32], src[33]);
      r2[1] = pk2h(src[34], src[35]);
      r2[2] = pk2h(src[36], src[37]);
      r2[3] = pk2h(src[38], 1.0f);   // ones column at K=39
    }
    sB1 = __builtin_bit_cast(v16h, r2);
  }

  v8f ct0 = {}, ct1 = {};            // cell state, D layout (h rows 0-15 / 16-31)

  // one gate D-tile: gates^T tile mt = W_ih[mt] @ [s;1]^T + W_hh[mt] @ h^T
  auto gate = [&](int mt) -> v8f {
    v16h w0 = wt[2 * mt][lane];      // ds_load_b128 x2, short live range
    v16h w1 = wt[2 * mt + 1][lane];
    v16h wh = wt[NT_IH + mt][lane];
    v8f acc = {};
    acc = wmma(w0, sB0, acc);
    acc = wmma(w1, sB1, acc);
    acc = wmma(wh, hB, acc);
    return acc;
  };

  for (int step = 0; step < pred_len; ++step) {
    v8f h0, h1;

    // ---- h rows 0-15: gate tiles i=0, f=2, g=4, o=6 ----
    {
      v8f gi = gate(0), gf = gate(2), gg = gate(4), go = gate(6);
#pragma unroll
      for (int r = 0; r < 8; ++r) {
        float iv = sigm(gi[r]), fv = sigm(gf[r]), gv = tanh_fast(gg[r]), ov = sigm(go[r]);
        ct0[r] = fv * ct0[r] + iv * gv;
        h0[r]  = ov * tanh_fast(ct0[r]);
      }
    }
    // ---- h rows 16-31: gate tiles i=1, f=3, g=5, o=7 ----
    {
      v8f gi = gate(1), gf = gate(3), gg = gate(5), go = gate(7);
#pragma unroll
      for (int r = 0; r < 8; ++r) {
        float iv = sigm(gi[r]), fv = sigm(gf[r]), gv = tanh_fast(gg[r]), ov = sigm(go[r]);
        ct1[r] = fv * ct1[r] + iv * gv;
        h1[r]  = ov * tanh_fast(ct1[r]);
      }
    }
    hB = DD_to_B(h0, h1, lane, bpidx);

    // ---- curr_s^T = W_mlp @ h^T + b_mlp  (3 M-tiles, pose rows valid 0-38) ----
    v16h wm0 = wt[24][lane], wm1 = wt[25][lane], wm2 = wt[26][lane];
    v8f s0 = wmma(wm0, hB, c_mlp[0]);
    v8f s1 = wmma(wm1, hB, c_mlp[1]);
    v8f s2 = wmma(wm2, hB, c_mlp[2]);

    // ---- store: out[step][col][p]; consecutive VGPRs = consecutive p -> b128 stores ----
    {
      float* ob = out + ((size_t)step * B + col) * 39;
      int ph = (lane < 16) ? 0 : 8;
      store4(ob + ph,          s0[0], s0[1], s0[2], s0[3]);
      store4(ob + ph + 4,      s0[4], s0[5], s0[6], s0[7]);
      store4(ob + 16 + ph,     s1[0], s1[1], s1[2], s1[3]);
      store4(ob + 16 + ph + 4, s1[4], s1[5], s1[6], s1[7]);
      if (lane < 16) {
        store4(ob + 32, s2[0], s2[1], s2[2], s2[3]);
        ob[36] = s2[4]; ob[37] = s2[5]; ob[38] = s2[6];   // p=39 is padding
      }
    }

    // ---- feed curr_s back as next step's B operand ----
    sB0 = DD_to_B(s0, s1, lane, bpidx);
    sB1 = D2_to_B(s2, bpidx);
  }
}

extern "C" void kernel_launch(void* const* d_in, const int* in_sizes, int n_in,
                              void* d_out, int out_size, void* d_ws, size_t ws_size,
                              hipStream_t stream) {
  const float* obs_s  = (const float*)d_in[0];
  const float* latent = (const float*)d_in[1];
  const float* W_ih   = (const float*)d_in[2];
  const float* W_hh   = (const float*)d_in[3];
  const float* b_ih   = (const float*)d_in[4];
  const float* b_hh   = (const float*)d_in[5];
  const float* W_fc   = (const float*)d_in[6];
  const float* b_fc   = (const float*)d_in[7];
  const float* W_mlp  = (const float*)d_in[8];
  const float* b_mlp  = (const float*)d_in[9];
  const int*   plen   = (const int*)d_in[10];

  int B       = in_sizes[1] / 16;          // latent is [B,16]
  int obs_len = in_sizes[0] / (B * 39);    // obs_s is [obs_len,B,39]
  int waves   = (B + 15) / 16;             // one wave per 16 batch rows
  int blocks  = (waves + 7) / 8;           // 8 waves (256 threads) per block

  lstm_decoder_kernel<<<blocks, 256, 0, stream>>>(
      obs_s, latent, W_ih, W_hh, b_ih, b_hh, W_fc, b_fc, W_mlp, b_mlp,
      plen, (float*)d_out, B, obs_len);
}
